// SlotAttention_62328565399751
// MI455X (gfx1250) — matrix operand order repaired and checked by reading, not compiled, not run on hardware
//
#include <hip/hip_runtime.h>

#define Bdim 32
#define Ndim 4096
#define Idim 512
#define Sdim 512
#define NSd  16
#define Hdim 1024

typedef __bf16 bf16;
typedef __attribute__((ext_vector_type(8)))  bf16  bf16x8;
typedef __attribute__((ext_vector_type(16))) bf16  v16bf;
typedef __attribute__((ext_vector_type(8)))  float v8f;

// ---------------- wave reductions (wave32) ----------------
__device__ inline float wave_sum(float v) {
#pragma unroll
  for (int m = 16; m >= 1; m >>= 1) v += __shfl_xor(v, m, 32);
  return v;
}
__device__ inline float half_sum(float v) {   // reduce across 16-lane half
#pragma unroll
  for (int m = 8; m >= 1; m >>= 1) v += __shfl_xor(v, m, 32);
  return v;
}
__device__ inline float half_max(float v) {
#pragma unroll
  for (int m = 8; m >= 1; m >>= 1) v = fmaxf(v, __shfl_xor(v, m, 32));
  return v;
}

// ---------------- WMMA fragment loaders (ISA 7.12.2 layouts) ----------------
// A operand (16xK, per-lane row m=lane&15): K pattern {kb..kb+7, kb+16..kb+23},
// kb = k0 + 8*(lane>=16)
__device__ inline v16bf fragA_bf16(const bf16* row, int k0) {
  const int lane = threadIdx.x & 31;
  const int kb = k0 + ((lane >> 4) << 3);
  bf16x8 lo = *(const bf16x8*)(row + kb);
  bf16x8 hi = *(const bf16x8*)(row + kb + 16);
  v16bf f;
#pragma unroll
  for (int t = 0; t < 8; ++t) { f[t] = lo[t]; f[t + 8] = hi[t]; }
  return f;
}
// B operand (Kx16, per-lane column n=lane&15): K pattern {kb..kb+7, kb+16..kb+23},
// kb = k0 + 8*(lane>=16)
__device__ inline v16bf fragB_bf16(const bf16* col, int k0) {
  const int lane = threadIdx.x & 31;
  const int kb = k0 + ((lane >> 4) << 3);
  bf16x8 lo = *(const bf16x8*)(col + kb);
  bf16x8 hi = *(const bf16x8*)(col + kb + 16);
  v16bf f;
#pragma unroll
  for (int t = 0; t < 8; ++t) { f[t] = lo[t]; f[t + 8] = hi[t]; }
  return f;
}
// A operand sourced from fp32 memory, converted on the fly
__device__ inline v16bf fragA_f32bf(const float* row, int k0) {
  const int lane = threadIdx.x & 31;
  const int kb = k0 + ((lane >> 4) << 3);
  v16bf f;
#pragma unroll
  for (int t = 0; t < 8; ++t) { f[t] = (bf16)row[kb + t]; f[t + 8] = (bf16)row[kb + 16 + t]; }
  return f;
}
__device__ inline v8f wmma_bf16(v16bf a, v16bf b, v8f c) {
  return __builtin_amdgcn_wmma_f32_16x16x32_bf16(false, a, false, b, (short)0, c, false, false);
}

// ---------------- kernels ----------------
__global__ void k_cvt_bf16(const float* __restrict__ src, bf16* __restrict__ dst, int n) {
  int i = blockIdx.x * 256 + threadIdx.x;
  if (i < n) dst[i] = (bf16)src[i];
}

__global__ void k_slots_init(const float* __restrict__ mu, const float* __restrict__ ls,
                             const float* __restrict__ noise, float* __restrict__ slots) {
  int i = blockIdx.x * 256 + threadIdx.x;
  if (i < Bdim * NSd * Sdim) {
    int js = i % (NSd * Sdim);
    slots[i] = mu[js] + __expf(ls[js]) * noise[i];
  }
}

// Fused LN(x) + keys/values projection.  Grid: B * N/16 blocks, 256 thr (8 waves).
// keys  : [B,N,S] bf16 (row-major, A-operand friendly)
// valsT : [B,S,N] bf16 (transposed, B-operand friendly for attn^T @ values)
__global__ __launch_bounds__(256) void k_ln_project(
    const float* __restrict__ x, const float* __restrict__ gam, const float* __restrict__ bet,
    const bf16* __restrict__ wk, const bf16* __restrict__ wv,
    bf16* __restrict__ keys, bf16* __restrict__ valsT) {
  __shared__ bf16 a_lds[16 * Idim];  // 16 KB
  const int b    = blockIdx.x >> 8;
  const int n0   = (blockIdx.x & 255) << 4;
  const int wave = threadIdx.x >> 5;
  const int lane = threadIdx.x & 31;

  for (int rr = 0; rr < 2; ++rr) {                 // 8 waves x 2 rows = 16 rows
    const int row = wave * 2 + rr;
    const float* xr = x + ((size_t)b * Ndim + n0 + row) * Idim;
    float vals[16];
    float s = 0.f;
#pragma unroll
    for (int t = 0; t < 16; ++t) { vals[t] = xr[lane + 32 * t]; s += vals[t]; }
    s = wave_sum(s);
    const float mean = s * (1.f / Idim);
    float vv = 0.f;
#pragma unroll
    for (int t = 0; t < 16; ++t) { float d = vals[t] - mean; vv += d * d; }
    vv = wave_sum(vv);
    const float rstd = rsqrtf(vv * (1.f / Idim) + 1e-5f);
#pragma unroll
    for (int t = 0; t < 16; ++t) {
      const int c = lane + 32 * t;
      a_lds[row * Idim + c] = (bf16)((vals[t] - mean) * rstd * gam[c] + bet[c]);
    }
  }
  __syncthreads();

  const int nn = lane & 15;
  const int mb = (lane >> 4) << 3;
  const bf16* arow = a_lds + nn * Idim;
  for (int tt = 0; tt < 8; ++tt) {                 // 64 col tiles (32 K + 32 V), 8/wave
    const int tile = wave * 8 + tt;
    const bool is_v = tile >= 32;
    const int s0 = (tile & 31) << 4;
    const bf16* wrow = (is_v ? wv : wk) + (size_t)(s0 + nn) * Idim;
    // 2-deep software pipeline on B (weights): keep two fragment loads in
    // flight so the scheduler emits partial loadcnt waits, not wait-to-zero.
    v16bf b0 = fragB_bf16(wrow, 0);
    v16bf b1 = fragB_bf16(wrow, 32);
    v8f c = {};
#pragma unroll 2
    for (int k0 = 0; k0 < Idim - 64; k0 += 32) {
      v16bf b2 = fragB_bf16(wrow, k0 + 64);
      c = wmma_bf16(fragA_bf16(arow, k0), b0, c);
      b0 = b1;
      b1 = b2;
    }
    c = wmma_bf16(fragA_bf16(arow, Idim - 64), b0, c);
    c = wmma_bf16(fragA_bf16(arow, Idim - 32), b1, c);
    if (!is_v) {
#pragma unroll
      for (int r = 0; r < 8; ++r)
        keys[((size_t)b * Ndim + n0 + mb + r) * Sdim + s0 + nn] = (bf16)c[r];
    } else {
#pragma unroll
      for (int r = 0; r < 8; ++r)
        valsT[((size_t)b * Sdim + s0 + nn) * Ndim + n0 + mb + r] = (bf16)c[r];
    }
  }
}

// q = LN(slots) @ Wq^T * S^-0.5   -> q bf16 [B,NS,S].  Grid: B blocks, 256 thr.
__global__ __launch_bounds__(256) void k_qproj(
    const float* __restrict__ slots, const float* __restrict__ gam, const float* __restrict__ bet,
    const bf16* __restrict__ wqb, bf16* __restrict__ qout) {
  __shared__ bf16 ln_s[NSd * Sdim];  // 16 KB
  const int b = blockIdx.x;
  const int wave = threadIdx.x >> 5, lane = threadIdx.x & 31;
  for (int rr = 0; rr < 2; ++rr) {
    const int row = wave * 2 + rr;
    const float* xr = slots + ((size_t)b * NSd + row) * Sdim;
    float vals[16];
    float s = 0.f;
#pragma unroll
    for (int t = 0; t < 16; ++t) { vals[t] = xr[lane + 32 * t]; s += vals[t]; }
    s = wave_sum(s);
    const float mean = s * (1.f / Sdim);
    float vv = 0.f;
#pragma unroll
    for (int t = 0; t < 16; ++t) { float d = vals[t] - mean; vv += d * d; }
    vv = wave_sum(vv);
    const float rstd = rsqrtf(vv * (1.f / Sdim) + 1e-5f);
#pragma unroll
    for (int t = 0; t < 16; ++t) {
      const int c = lane + 32 * t;
      ln_s[row * Sdim + c] = (bf16)((vals[t] - mean) * rstd * gam[c] + bet[c]);
    }
  }
  __syncthreads();
  const float scale = 0.04419417382415922f;  // 512^-0.5
  const int nn = lane & 15, mb = (lane >> 4) << 3;
  const bf16* ar = ln_s + nn * Sdim;
  for (int tt = 0; tt < 4; ++tt) {             // 32 tiles / 8 waves
    const int s0 = ((wave << 2) + tt) << 4;
    const bf16* br = wqb + (size_t)(s0 + nn) * Sdim;
    v8f c = {};
    for (int k0 = 0; k0 < Sdim; k0 += 32)
      c = wmma_bf16(fragA_bf16(ar, k0), fragB_bf16(br, k0), c);
#pragma unroll
    for (int r = 0; r < 8; ++r)
      qout[((size_t)b * NSd + mb + r) * Sdim + s0 + nn] = (bf16)(c[r] * scale);
  }
}

// logits = keys @ q^T, softmax over the 16 slots, +eps, column-sum -> den.
// q tile staged in LDS; keys stream 2-deep pipelined from HBM.
// Stores attn transposed: attn_t[b, j, i].  Grid: B * N/128 blocks, 256 thr.
__global__ __launch_bounds__(256) void k_attn(
    const bf16* __restrict__ keys, const bf16* __restrict__ qv,
    float* __restrict__ attn_t, float* __restrict__ den) {
  __shared__ bf16 q_l[NSd * Sdim];  // 16 KB
  __shared__ float den_l[NSd];
  const int b  = blockIdx.x >> 5;
  const int i0 = (blockIdx.x & 31) * 128;
  const int wave = threadIdx.x >> 5, lane = threadIdx.x & 31;
  {
    const bf16x8* src = (const bf16x8*)(qv + (size_t)b * NSd * Sdim);
    bf16x8* dst = (bf16x8*)q_l;
    for (int t = threadIdx.x; t < NSd * Sdim / 8; t += 256) dst[t] = src[t];
  }
  if (threadIdx.x < NSd) den_l[threadIdx.x] = 0.f;
  __syncthreads();

  const int it0 = i0 + wave * 16;
  const int nn = lane & 15, mb = (lane >> 4) << 3;
  const bf16* krow = keys + ((size_t)b * Ndim + it0 + nn) * Sdim;
  const bf16* qrow = q_l + nn * Sdim;
  v16bf a0 = fragA_bf16(krow, 0);
  v16bf a1 = fragA_bf16(krow, 32);
  v8f c = {};
#pragma unroll 2
  for (int k0 = 0; k0 < Sdim - 64; k0 += 32) {
    v16bf a2 = fragA_bf16(krow, k0 + 64);
    __builtin_prefetch(krow + k0 + 512, 0, 3);
    c = wmma_bf16(a0, fragB_bf16(qrow, k0), c);
    a0 = a1;
    a1 = a2;
  }
  c = wmma_bf16(a0, fragB_bf16(qrow, Sdim - 64), c);
  c = wmma_bf16(a1, fragB_bf16(qrow, Sdim - 32), c);

  // softmax over j (16 lanes of each half-wave), + eps
  float lsum = 0.f;
#pragma unroll
  for (int r = 0; r < 8; ++r) {
    float v = c[r];
    float mx = half_max(v);
    float e = __expf(v - mx);
    float se = half_sum(e);
    float a = e / se + 1e-8f;
    c[r] = a;
    lsum += a;
  }
  atomicAdd(&den_l[nn], lsum);
  float* arow = attn_t + ((size_t)b * NSd + nn) * Ndim;
#pragma unroll
  for (int r = 0; r < 8; ++r) arow[it0 + mb + r] = c[r];
  __syncthreads();
  if (threadIdx.x < NSd) atomicAdd(&den[b * NSd + threadIdx.x], den_l[threadIdx.x]);
}

// updates[b,j,s] = (1/den[b,j]) * sum_i attn_t[b,j,i] * values[b,i,s]
// Grid: B*4 blocks (128 cols each), 256 thr; K = N = 4096.  A and B 2-deep pipelined.
__global__ __launch_bounds__(256) void k_updates(
    const float* __restrict__ attn_t, const bf16* __restrict__ valsT,
    const float* __restrict__ den, float* __restrict__ upd) {
  const int b = blockIdx.x >> 2;
  const int s0 = (blockIdx.x & 3) * 128 + (threadIdx.x >> 5) * 16;
  const int lane = threadIdx.x & 31;
  const int nn = lane & 15, mb = (lane >> 4) << 3;
  const float* arow = attn_t + ((size_t)b * NSd + nn) * Ndim;
  const bf16* vrow = valsT + ((size_t)b * Sdim + s0 + nn) * Ndim;
  v16bf a0 = fragA_f32bf(arow, 0);
  v16bf b0 = fragB_bf16(vrow, 0);
  v16bf a1 = fragA_f32bf(arow, 32);
  v16bf b1 = fragB_bf16(vrow, 32);
  v8f c = {};
#pragma unroll 2
  for (int k0 = 0; k0 < Ndim - 64; k0 += 32) {
    v16bf a2 = fragA_f32bf(arow, k0 + 64);
    v16bf b2 = fragB_bf16(vrow, k0 + 64);
    __builtin_prefetch(vrow + k0 + 512, 0, 3);
    c = wmma_bf16(a0, b0, c);
    a0 = a1; a1 = a2;
    b0 = b1; b1 = b2;
  }
  c = wmma_bf16(a0, b0, c);
  c = wmma_bf16(a1, b1, c);
#pragma unroll
  for (int r = 0; r < 8; ++r) {
    const int j = mb + r;
    upd[((size_t)b * NSd + j) * Sdim + s0 + nn] = c[r] / den[b * NSd + j];
  }
}

// Fused GRUCell (r,z,n gate ordering).  Grid: B blocks, 512 thr (16 waves).
__global__ __launch_bounds__(512) void k_gru(
    const float* __restrict__ upd, const float* __restrict__ slots_in,
    const bf16* __restrict__ wihb, const bf16* __restrict__ whhb,
    const float* __restrict__ bih, const float* __restrict__ bhh,
    float* __restrict__ out) {
  __shared__ bf16 u_l[NSd * Sdim];  // 16 KB
  __shared__ bf16 h_l[NSd * Sdim];  // 16 KB
  const int b = blockIdx.x;
  const float* ub = upd + (size_t)b * NSd * Sdim;
  const float* hb = slots_in + (size_t)b * NSd * Sdim;
  for (int t = threadIdx.x; t < NSd * Sdim; t += 512) {
    u_l[t] = (bf16)ub[t];
    h_l[t] = (bf16)hb[t];
  }
  __syncthreads();
  const int wave = threadIdx.x >> 5, lane = threadIdx.x & 31;
  const int nn = lane & 15, mb = (lane >> 4) << 3;
  const bf16* au = u_l + nn * Sdim;
  const bf16* ah = h_l + nn * Sdim;
  for (int tt = 0; tt < 2; ++tt) {                  // 32 s-tiles / 16 waves
    const int s0 = ((wave << 1) + tt) << 4;
    const bf16* wr = wihb + (size_t)(0 * Sdim + s0 + nn) * Sdim;
    const bf16* wz = wihb + (size_t)(1 * Sdim + s0 + nn) * Sdim;
    const bf16* wn = wihb + (size_t)(2 * Sdim + s0 + nn) * Sdim;
    const bf16* vr = whhb + (size_t)(0 * Sdim + s0 + nn) * Sdim;
    const bf16* vz = whhb + (size_t)(1 * Sdim + s0 + nn) * Sdim;
    const bf16* vn = whhb + (size_t)(2 * Sdim + s0 + nn) * Sdim;
    v8f xr = {}, xz = {}, xn = {}, hr = {}, hz = {}, hn = {};
    for (int k0 = 0; k0 < Sdim; k0 += 32) {
      v16bf aU = fragA_bf16(au, k0);
      v16bf aH = fragA_bf16(ah, k0);
      xr = wmma_bf16(aU, fragB_bf16(wr, k0), xr);
      xz = wmma_bf16(aU, fragB_bf16(wz, k0), xz);
      xn = wmma_bf16(aU, fragB_bf16(wn, k0), xn);
      hr = wmma_bf16(aH, fragB_bf16(vr, k0), hr);
      hz = wmma_bf16(aH, fragB_bf16(vz, k0), hz);
      hn = wmma_bf16(aH, fragB_bf16(vn, k0), hn);
    }
    const int col = s0 + nn;
    const float bir = bih[col], biz = bih[Sdim + col], bin = bih[2 * Sdim + col];
    const float bhr = bhh[col], bhz = bhh[Sdim + col], bhn = bhh[2 * Sdim + col];
#pragma unroll
    for (int r = 0; r < 8; ++r) {
      const int m = mb + r;
      const float rg = 1.f / (1.f + __expf(-(xr[r] + bir + hr[r] + bhr)));
      const float zg = 1.f / (1.f + __expf(-(xz[r] + biz + hz[r] + bhz)));
      const float ng = tanhf(xn[r] + bin + rg * (hn[r] + bhn));
      const size_t idx = ((size_t)b * NSd + m) * Sdim + col;
      out[idx] = (1.f - zg) * ng + zg * slots_in[idx];
    }
  }
}

// slots = updates + MLP(LN(updates)).  Grid: B blocks, 512 thr (16 waves).
__global__ __launch_bounds__(512) void k_mlp(
    const float* __restrict__ updg, const float* __restrict__ gam, const float* __restrict__ bet,
    const bf16* __restrict__ w1b, const float* __restrict__ b1v,
    const bf16* __restrict__ w2b, const float* __restrict__ b2v,
    float* __restrict__ slots_out) {
  __shared__ bf16 ln_l[NSd * Sdim];  // 16 KB
  __shared__ bf16 h_l[NSd * Hdim];   // 32 KB
  const int b = blockIdx.x;
  const int wave = threadIdx.x >> 5, lane = threadIdx.x & 31;
  {  // LN: one row per wave
    const int row = wave;
    const float* xr = updg + ((size_t)b * NSd + row) * Sdim;
    float vals[16];
    float s = 0.f;
#pragma unroll
    for (int t = 0; t < 16; ++t) { vals[t] = xr[lane + 32 * t]; s += vals[t]; }
    s = wave_sum(s);
    const float mean = s * (1.f / Sdim);
    float vv = 0.f;
#pragma unroll
    for (int t = 0; t < 16; ++t) { float d = vals[t] - mean; vv += d * d; }
    vv = wave_sum(vv);
    const float rstd = rsqrtf(vv * (1.f / Sdim) + 1e-5f);
#pragma unroll
    for (int t = 0; t < 16; ++t) {
      const int c = lane + 32 * t;
      ln_l[row * Sdim + c] = (bf16)((vals[t] - mean) * rstd * gam[c] + bet[c]);
    }
  }
  __syncthreads();
  const int nn = lane & 15, mb = (lane >> 4) << 3;
  const bf16* ar = ln_l + nn * Sdim;
  for (int tt = 0; tt < 4; ++tt) {                  // 64 h-tiles / 16 waves
    const int h0 = ((wave << 2) + tt) << 4;
    const bf16* br = w1b + (size_t)(h0 + nn) * Sdim;
    v8f c = {};
    for (int k0 = 0; k0 < Sdim; k0 += 32)
      c = wmma_bf16(fragA_bf16(ar, k0), fragB_bf16(br, k0), c);
    const float bb = b1v[h0 + nn];
#pragma unroll
    for (int r = 0; r < 8; ++r)
      h_l[(mb + r) * Hdim + h0 + nn] = (bf16)fmaxf(c[r] + bb, 0.f);
  }
  __syncthreads();
  const bf16* ar2 = h_l + nn * Hdim;
  for (int tt = 0; tt < 2; ++tt) {                  // 32 s-tiles / 16 waves
    const int s0 = ((wave << 1) + tt) << 4;
    const bf16* br = w2b + (size_t)(s0 + nn) * Hdim;
    v8f c = {};
    for (int k0 = 0; k0 < Hdim; k0 += 32)
      c = wmma_bf16(fragA_bf16(ar2, k0), fragB_bf16(br, k0), c);
    const float bb = b2v[s0 + nn];
#pragma unroll
    for (int r = 0; r < 8; ++r) {
      const size_t idx = ((size_t)b * NSd + mb + r) * Sdim + s0 + nn;
      slots_out[idx] = updg[idx] + c[r] + bb;
    }
  }
}

// Final normalized attention, back to [B, N, NS] layout.
__global__ void k_attn_out(const float* __restrict__ attn_t, const float* __restrict__ den,
                           float* __restrict__ out) {
  const size_t i = (size_t)blockIdx.x * 256 + threadIdx.x;
  if (i < (size_t)Bdim * Ndim * NSd) {
    const int j = (int)(i % NSd);
    const size_t rem = i / NSd;
    const int ii = (int)(rem % Ndim);
    const int b = (int)(rem / Ndim);
    out[i] = attn_t[((size_t)b * NSd + j) * Ndim + ii] / den[b * NSd + j];
  }
}

// ---------------- host side ----------------
extern "C" void kernel_launch(void* const* d_in, const int* in_sizes, int n_in,
                              void* d_out, int out_size, void* d_ws, size_t ws_size,
                              hipStream_t stream) {
  (void)in_sizes; (void)n_in; (void)out_size; (void)ws_size;
  const float* x       = (const float*)d_in[0];
  const float* noise   = (const float*)d_in[1];
  const float* ln_in_g = (const float*)d_in[2];
  const float* ln_in_b = (const float*)d_in[3];
  const float* ln_sl_g = (const float*)d_in[4];
  const float* ln_sl_b = (const float*)d_in[5];
  const float* ln_ml_g = (const float*)d_in[6];
  const float* ln_ml_b = (const float*)d_in[7];
  const float* mu      = (const float*)d_in[8];
  const float* lsig    = (const float*)d_in[9];
  const float* Wq      = (const float*)d_in[10];
  const float* Wk      = (const float*)d_in[11];
  const float* Wv      = (const float*)d_in[12];
  const float* Wih     = (const float*)d_in[13];
  const float* Whh     = (const float*)d_in[14];
  const float* bih     = (const float*)d_in[15];
  const float* bhh     = (const float*)d_in[16];
  const float* W1      = (const float*)d_in[17];
  const float* b1      = (const float*)d_in[18];
  const float* W2      = (const float*)d_in[19];
  const float* b2      = (const float*)d_in[20];

  char* ws = (char*)d_ws;
  size_t off = 0;
  auto alloc = [&](size_t bytes) -> void* {
    void* p = ws + off;
    off += (bytes + 255) & ~(size_t)255;
    return p;
  };
  bf16* keys   = (bf16*)alloc((size_t)Bdim * Ndim * Sdim * 2);
  bf16* valsT  = (bf16*)alloc((size_t)Bdim * Ndim * Sdim * 2);
  bf16* wk_b   = (bf16*)alloc((size_t)Sdim * Idim * 2);
  bf16* wv_b   = (bf16*)alloc((size_t)Sdim * Idim * 2);
  bf16* wq_b   = (bf16*)alloc((size_t)Sdim * Sdim * 2);
  bf16* wih_b  = (bf16*)alloc((size_t)3 * Sdim * Sdim * 2);
  bf16* whh_b  = (bf16*)alloc((size_t)3 * Sdim * Sdim * 2);
  bf16* w1_b   = (bf16*)alloc((size_t)Hdim * Sdim * 2);
  bf16* w2_b   = (bf16*)alloc((size_t)Sdim * Hdim * 2);
  bf16* qb     = (bf16*)alloc((size_t)Bdim * NSd * Sdim * 2);
  float* attn_t = (float*)alloc((size_t)Bdim * NSd * Ndim * 4);
  float* den    = (float*)alloc((size_t)Bdim * NSd * 4);
  float* slots  = (float*)alloc((size_t)Bdim * NSd * Sdim * 4);
  float* upd    = (float*)alloc((size_t)Bdim * NSd * Sdim * 4);
  float* updg   = (float*)alloc((size_t)Bdim * NSd * Sdim * 4);

  auto cvt = [&](const float* s, bf16* d, int n) {
    k_cvt_bf16<<<(n + 255) / 256, 256, 0, stream>>>(s, d, n);
  };
  cvt(Wk, wk_b, Sdim * Idim);
  cvt(Wv, wv_b, Sdim * Idim);
  cvt(Wq, wq_b, Sdim * Sdim);
  cvt(Wih, wih_b, 3 * Sdim * Sdim);
  cvt(Whh, whh_b, 3 * Sdim * Sdim);
  cvt(W1, w1_b, Hdim * Sdim);
  cvt(W2, w2_b, Sdim * Hdim);

  k_ln_project<<<Bdim * (Ndim / 16), 256, 0, stream>>>(x, ln_in_g, ln_in_b, wk_b, wv_b,
                                                       keys, valsT);
  k_slots_init<<<(Bdim * NSd * Sdim + 255) / 256, 256, 0, stream>>>(mu, lsig, noise, slots);

  for (int it = 0; it < 3; ++it) {
    hipMemsetAsync(den, 0, (size_t)Bdim * NSd * 4, stream);
    k_qproj<<<Bdim, 256, 0, stream>>>(slots, ln_sl_g, ln_sl_b, wq_b, qb);
    k_attn<<<Bdim * (Ndim / 128), 256, 0, stream>>>(keys, qb, attn_t, den);
    k_updates<<<Bdim * 4, 256, 0, stream>>>(attn_t, valsT, den, upd);
    k_gru<<<Bdim, 512, 0, stream>>>(upd, slots, wih_b, whh_b, bih, bhh, updg);
    k_mlp<<<Bdim, 512, 0, stream>>>(updg, ln_ml_g, ln_ml_b, w1_b, b1, w2_b, b2, slots);
  }

  // outputs: slots [B,NS,S] then attn [B,N,NS]
  hipMemcpyAsync(d_out, slots, (size_t)Bdim * NSd * Sdim * 4, hipMemcpyDeviceToDevice, stream);
  k_attn_out<<<(Bdim * Ndim * NSd + 255) / 256, 256, 0, stream>>>(
      attn_t, den, (float*)d_out + (size_t)Bdim * NSd * Sdim);
}
